// PathologySpecificSequenceAttention_21131239097005
// MI455X (gfx1250) — matrix operand order repaired and checked by reading, not compiled, not run on hardware
//
#include <hip/hip_runtime.h>
#include <hip/hip_bf16.h>

// ---------------------------------------------------------------------------
// Problem constants
// ---------------------------------------------------------------------------
#define BATCH   8192
#define SEQ     4
#define DMODEL  2048
#define NHEAD   16
#define HEADD   128           // DMODEL / NHEAD
#define LN_EPS  1e-5f

// GEMM tiling: 256x128 block tile, 8 waves as 4(M) x 2(N), 64x64 per wave.
#define BM 256
#define BN 128
#define BK 32
#define LPAD 8                // bf16 elems of LDS row padding (keeps 16B align, spreads banks)

typedef __attribute__((ext_vector_type(16))) __bf16 bf16x16;
typedef __attribute__((ext_vector_type(8)))  float  f32x8;

union FragAB { bf16x16 v; uint4 q[2]; };

// ---------------------------------------------------------------------------
// bf16 <-> f32 helpers (RNE pack; unpack via bit shift)
// ---------------------------------------------------------------------------
__device__ __forceinline__ float bfbits2f(unsigned int u16) {
    union { unsigned int i; float f; } c;
    c.i = u16 << 16;
    return c.f;
}
__device__ __forceinline__ unsigned short f2bfbits(float f) {
    union { float f; unsigned int i; } c;
    c.f = f;
    unsigned int u = c.i;
    u += 0x7FFFu + ((u >> 16) & 1u);   // round-to-nearest-even
    return (unsigned short)(u >> 16);
}

// ---------------------------------------------------------------------------
// Kernel 1: cast + transpose weights to bf16  WT[n][k] = W[k][n]
//   blockIdx.z selects matrix: 0 = Wq[pathology_idx], 1 = Wk, 2 = Wv, 3 = Wo
// ---------------------------------------------------------------------------
__global__ __launch_bounds__(256)
void cast_weights_kernel(const float* __restrict__ Wq, const float* __restrict__ Wk,
                         const float* __restrict__ Wv, const float* __restrict__ Wo,
                         const int* __restrict__ pidx,
                         unsigned short* __restrict__ WqT, unsigned short* __restrict__ WkT,
                         unsigned short* __restrict__ WvT, unsigned short* __restrict__ WoT)
{
    __shared__ float tile[32][33];
    const float* src;
    unsigned short* dst;
    switch (blockIdx.z) {
        case 0:  src = Wq + (size_t)(*pidx) * DMODEL * DMODEL; dst = WqT; break;
        case 1:  src = Wk; dst = WkT; break;
        case 2:  src = Wv; dst = WvT; break;
        default: src = Wo; dst = WoT; break;
    }
    const int k0 = blockIdx.y * 32;
    const int n0 = blockIdx.x * 32;
    const int tx = threadIdx.x;      // 0..31
    const int ty = threadIdx.y;      // 0..7
#pragma unroll
    for (int j = 0; j < 4; ++j)
        tile[ty + j * 8][tx] = src[(size_t)(k0 + ty + j * 8) * DMODEL + n0 + tx];
    __syncthreads();
#pragma unroll
    for (int j = 0; j < 4; ++j)
        dst[(size_t)(n0 + ty + j * 8) * DMODEL + k0 + tx] = f2bfbits(tile[tx][ty + j * 8]);
}

// ---------------------------------------------------------------------------
// Kernel 2: mean over S + bf16 cast of seq_feats
// ---------------------------------------------------------------------------
__global__ __launch_bounds__(256)
void mean_cast_kernel(const float* __restrict__ seq,
                      unsigned short* __restrict__ seq_bf,
                      unsigned short* __restrict__ q_bf)
{
    const size_t idx = (size_t)blockIdx.x * 256 + threadIdx.x;  // over B*D
    const size_t b = idx / DMODEL;
    const size_t d = idx % DMODEL;
    float acc = 0.0f;
#pragma unroll
    for (int s = 0; s < SEQ; ++s) {
        const size_t off = (b * SEQ + s) * DMODEL + d;
        float v = seq[off];
        seq_bf[off] = f2bfbits(v);
        acc += v;
    }
    q_bf[idx] = f2bfbits(acc * (1.0f / SEQ));
}

// ---------------------------------------------------------------------------
// Kernel 3: bf16 WMMA GEMM   C[M,N] = A[M,K] * W[K,N] + bias
//   A row-major bf16, WT = W^T row-major bf16 (WT[n][k]), K = N = 2048.
//   Block: 256x128 tile, 256 threads (8 waves as 4(M) x 2(N)), BK=32,
//   double-buffered LDS, v_wmma_f32_16x16x32_bf16, 4x4 16x16 tiles per wave
//   (16 WMMAs per K-step vs 16 ds_load_b128 -> XDL-issue limited).
//   bias offset: pidx ? (*pidx)*bias_stride : 0  (for pathology-indexed bq)
// ---------------------------------------------------------------------------
template <bool OUT_F32>
__global__ __launch_bounds__(256)
void gemm_bf16_wmma(const unsigned short* __restrict__ A,
                    const unsigned short* __restrict__ WT,
                    const float* __restrict__ bias,
                    const int* __restrict__ pidx, int bias_stride,
                    void* __restrict__ Cout)
{
    __shared__ __align__(16) unsigned short sA[2][BM][BK + LPAD];   // 2*256*40*2 = 40 KB
    __shared__ __align__(16) unsigned short sB[2][BN][BK + LPAD];   // 2*128*40*2 = 20 KB

    const int tid   = threadIdx.x;
    const int wave  = tid >> 5;
    const int lane  = tid & 31;
    const int half  = lane >> 4;          // 0: lanes 0-15, 1: lanes 16-31
    const int r     = lane & 15;
    const int waveM = wave & 3;           // 4 waves along M -> 64 rows each
    const int waveN = wave >> 2;          // 2 waves along N -> 64 cols each
    const long blockM = (long)blockIdx.y * BM;
    const long blockN = (long)blockIdx.x * BN;

    const int boff = (pidx != nullptr) ? ((*pidx) * bias_stride) : 0;

    f32x8 acc[4][4];
#pragma unroll
    for (int mt = 0; mt < 4; ++mt)
#pragma unroll
        for (int nt = 0; nt < 4; ++nt) {
            f32x8 z = {0.f, 0.f, 0.f, 0.f, 0.f, 0.f, 0.f, 0.f};
            acc[mt][nt] = z;
        }

    const int nTiles = DMODEL / BK;   // 64
    uint4 ra[4], rb[2];

    // preload tile 0 into buffer 0
    // A tile: 256x32 bf16 = 1024 x 16B chunks -> 4 per thread
    // B tile: 128x32 bf16 =  512 x 16B chunks -> 2 per thread
#pragma unroll
    for (int i = 0; i < 4; ++i) {
        const int l = tid + i * 256;          // 0..1023
        const int row = l >> 2;               // 0..255
        const int col = (l & 3) * 8;          // 0,8,16,24
        ra[i] = *(const uint4*)(A + (size_t)(blockM + row) * DMODEL + col);
        *(uint4*)&sA[0][row][col] = ra[i];
    }
#pragma unroll
    for (int i = 0; i < 2; ++i) {
        const int l = tid + i * 256;          // 0..511
        const int row = l >> 2;               // 0..127
        const int col = (l & 3) * 8;
        rb[i] = *(const uint4*)(WT + (size_t)(blockN + row) * DMODEL + col);
        *(uint4*)&sB[0][row][col] = rb[i];
    }
    __syncthreads();

    for (int kt = 0; kt < nTiles; ++kt) {
        const int cur = kt & 1;
        const int nxt = cur ^ 1;
        const bool hasNext = (kt + 1) < nTiles;

        if (hasNext) {
            const int k0 = (kt + 1) * BK;
#pragma unroll
            for (int i = 0; i < 4; ++i) {
                const int l = tid + i * 256;
                const int row = l >> 2;
                const int col = (l & 3) * 8;
                ra[i] = *(const uint4*)(A + (size_t)(blockM + row) * DMODEL + k0 + col);
            }
#pragma unroll
            for (int i = 0; i < 2; ++i) {
                const int l = tid + i * 256;
                const int row = l >> 2;
                const int col = (l & 3) * 8;
                rb[i] = *(const uint4*)(WT + (size_t)(blockN + row) * DMODEL + k0 + col);
            }
        }

        // --- A fragments (kept live across the nt loop for reuse) ---
        FragAB fa[4];
#pragma unroll
        for (int mt = 0; mt < 4; ++mt) {
            const int row = waveM * 64 + mt * 16 + r;       // A rows: M = r (+tile)
            fa[mt].q[0] = *(const uint4*)&sA[cur][row][half * 8];        // K = half*8 + 0..7
            fa[mt].q[1] = *(const uint4*)&sA[cur][row][16 + half * 8];   // K = 16+half*8 + 0..7
        }
        // --- B fragment loaded per nt to keep live ranges short ---
#pragma unroll
        for (int nt = 0; nt < 4; ++nt) {
            FragAB fb;
            const int col = waveN * 64 + nt * 16 + r;       // B cols: N = r (+tile)
            fb.q[0] = *(const uint4*)&sB[cur][col][half * 8];            // K = half*8 + 0..7
            fb.q[1] = *(const uint4*)&sB[cur][col][16 + half * 8];       // K = 16+half*8 + 0..7
#pragma unroll
            for (int mt = 0; mt < 4; ++mt)
                acc[mt][nt] = __builtin_amdgcn_wmma_f32_16x16x32_bf16(
                    false, fa[mt].v, false, fb.v,
                    (short)0, acc[mt][nt], false, false);
        }

        if (hasNext) {
#pragma unroll
            for (int i = 0; i < 4; ++i) {
                const int l = tid + i * 256;
                const int row = l >> 2;
                const int col = (l & 3) * 8;
                *(uint4*)&sA[nxt][row][col] = ra[i];
            }
#pragma unroll
            for (int i = 0; i < 2; ++i) {
                const int l = tid + i * 256;
                const int row = l >> 2;
                const int col = (l & 3) * 8;
                *(uint4*)&sB[nxt][row][col] = rb[i];
            }
        }
        __syncthreads();
    }

    // --- epilogue: C layout VGPR i -> row = half*8 + i, col = r (per tile) ---
#pragma unroll
    for (int mt = 0; mt < 4; ++mt) {
#pragma unroll
        for (int nt = 0; nt < 4; ++nt) {
            const long col = blockN + waveN * 64 + nt * 16 + r;
            const float bv = bias[boff + col];
#pragma unroll
            for (int i = 0; i < 8; ++i) {
                const long row = blockM + waveM * 64 + mt * 16 + half * 8 + i;
                const float v = acc[mt][nt][i] + bv;
                if (OUT_F32)
                    ((float*)Cout)[(size_t)row * DMODEL + col] = v;
                else
                    ((unsigned short*)Cout)[(size_t)row * DMODEL + col] = f2bfbits(v);
            }
        }
    }
}

// ---------------------------------------------------------------------------
// Kernel 4: attention — one wave per (b, h). S=4, HD=128 (4 elems/lane).
// ---------------------------------------------------------------------------
__global__ __launch_bounds__(256)
void attention_kernel(const unsigned short* __restrict__ Qb,
                      const unsigned short* __restrict__ Kb,
                      const unsigned short* __restrict__ Vb,
                      const unsigned char* __restrict__ avail,
                      unsigned short* __restrict__ ctx)
{
    const int wave = threadIdx.x >> 5;
    const int lane = threadIdx.x & 31;
    const int pair = blockIdx.x * 8 + wave;   // b*H + h
    const int b = pair >> 4;                  // H = 16
    const int h = pair & 15;

    const size_t qoff = (size_t)b * DMODEL + h * HEADD + lane * 4;
    const uint2 qv = *(const uint2*)(Qb + qoff);
    float q0 = bfbits2f(qv.x & 0xffffu), q1 = bfbits2f(qv.x >> 16);
    float q2 = bfbits2f(qv.y & 0xffffu), q3 = bfbits2f(qv.y >> 16);

    float score[SEQ];
#pragma unroll
    for (int s = 0; s < SEQ; ++s) {
        const size_t koff = ((size_t)(b * SEQ + s)) * DMODEL + h * HEADD + lane * 4;
        const uint2 kv = *(const uint2*)(Kb + koff);
        float p = q0 * bfbits2f(kv.x & 0xffffu) + q1 * bfbits2f(kv.x >> 16)
                + q2 * bfbits2f(kv.y & 0xffffu) + q3 * bfbits2f(kv.y >> 16);
#pragma unroll
        for (int off = 16; off > 0; off >>= 1) p += __shfl_xor(p, off, 32);
        score[s] = p * 0.088388347648318447f;  // 1/sqrt(128)
    }

    bool av[SEQ];
    float m = -INFINITY;
#pragma unroll
    for (int s = 0; s < SEQ; ++s) {
        av[s] = avail[b * SEQ + s] != 0;
        if (av[s]) m = fmaxf(m, score[s]);
    }
    float e[SEQ], sum = 0.0f;
#pragma unroll
    for (int s = 0; s < SEQ; ++s) {
        e[s] = av[s] ? __expf(score[s] - m) : 0.0f;
        sum += e[s];
    }
    const float inv = 1.0f / sum;

    float c0 = 0.f, c1 = 0.f, c2 = 0.f, c3 = 0.f;
#pragma unroll
    for (int s = 0; s < SEQ; ++s) {
        const float a = e[s] * inv;
        const size_t voff = ((size_t)(b * SEQ + s)) * DMODEL + h * HEADD + lane * 4;
        const uint2 vv = *(const uint2*)(Vb + voff);
        c0 += a * bfbits2f(vv.x & 0xffffu); c1 += a * bfbits2f(vv.x >> 16);
        c2 += a * bfbits2f(vv.y & 0xffffu); c3 += a * bfbits2f(vv.y >> 16);
    }
    uint2 ov;
    ov.x = (unsigned int)f2bfbits(c0) | ((unsigned int)f2bfbits(c1) << 16);
    ov.y = (unsigned int)f2bfbits(c2) | ((unsigned int)f2bfbits(c3) << 16);
    *(uint2*)(ctx + qoff) = ov;
}

// ---------------------------------------------------------------------------
// Kernel 5: LayerNorm in place on d_out — one block per row of 2048.
// ---------------------------------------------------------------------------
__global__ __launch_bounds__(256)
void layernorm_kernel(float* __restrict__ out,
                      const float* __restrict__ gamma,
                      const float* __restrict__ beta)
{
    __shared__ float redS[8], redQ[8];
    float* p = out + (size_t)blockIdx.x * DMODEL;
    float vals[8], sum = 0.f, sq = 0.f;
#pragma unroll
    for (int i = 0; i < 8; ++i) {
        float v = p[threadIdx.x + i * 256];
        vals[i] = v; sum += v; sq += v * v;
    }
#pragma unroll
    for (int off = 16; off > 0; off >>= 1) {
        sum += __shfl_xor(sum, off, 32);
        sq  += __shfl_xor(sq,  off, 32);
    }
    const int wave = threadIdx.x >> 5, lane = threadIdx.x & 31;
    if (lane == 0) { redS[wave] = sum; redQ[wave] = sq; }
    __syncthreads();
    sum = 0.f; sq = 0.f;
#pragma unroll
    for (int i = 0; i < 8; ++i) { sum += redS[i]; sq += redQ[i]; }
    const float mu  = sum * (1.0f / DMODEL);
    const float var = sq  * (1.0f / DMODEL) - mu * mu;
    const float rinv = rsqrtf(var + LN_EPS);
#pragma unroll
    for (int i = 0; i < 8; ++i) {
        const int c = threadIdx.x + i * 256;
        p[c] = (vals[i] - mu) * rinv * gamma[c] + beta[c];
    }
}

// ---------------------------------------------------------------------------
// Workspace layout (bytes) — total 512 MB
// ---------------------------------------------------------------------------
static const size_t OFF_SEQBF = 0;                                      // [B*S, D] bf16  128 MB
static const size_t OFF_QIN   = OFF_SEQBF + (size_t)BATCH*SEQ*DMODEL*2; // [B, D] bf16     32 MB
static const size_t OFF_WQT   = OFF_QIN   + (size_t)BATCH*DMODEL*2;     // 8 MB each
static const size_t OFF_WKT   = OFF_WQT   + (size_t)DMODEL*DMODEL*2;
static const size_t OFF_WVT   = OFF_WKT   + (size_t)DMODEL*DMODEL*2;
static const size_t OFF_WOT   = OFF_WVT   + (size_t)DMODEL*DMODEL*2;
static const size_t OFF_QB    = OFF_WOT   + (size_t)DMODEL*DMODEL*2;    // [B, D] bf16     32 MB
static const size_t OFF_KB    = OFF_QB    + (size_t)BATCH*DMODEL*2;     // [B*S, D]       128 MB
static const size_t OFF_VB    = OFF_KB    + (size_t)BATCH*SEQ*DMODEL*2; // [B*S, D]       128 MB
static const size_t OFF_CTX   = OFF_VB    + (size_t)BATCH*SEQ*DMODEL*2; // [B, D] bf16     32 MB

extern "C" void kernel_launch(void* const* d_in, const int* in_sizes, int n_in,
                              void* d_out, int out_size, void* d_ws, size_t ws_size,
                              hipStream_t stream) {
    const float*         seq   = (const float*)d_in[0];
    const unsigned char* avail = (const unsigned char*)d_in[1];
    const int*           pidx  = (const int*)d_in[2];
    const float* Wq = (const float*)d_in[3];
    const float* bq = (const float*)d_in[4];
    const float* Wk = (const float*)d_in[5];
    const float* bk = (const float*)d_in[6];
    const float* Wv = (const float*)d_in[7];
    const float* bv = (const float*)d_in[8];
    const float* Wo = (const float*)d_in[9];
    const float* bo = (const float*)d_in[10];
    const float* gamma = (const float*)d_in[11];
    const float* beta  = (const float*)d_in[12];
    float* out = (float*)d_out;

    char* ws = (char*)d_ws;
    unsigned short* seq_bf = (unsigned short*)(ws + OFF_SEQBF);
    unsigned short* q_bf   = (unsigned short*)(ws + OFF_QIN);
    unsigned short* WqT    = (unsigned short*)(ws + OFF_WQT);
    unsigned short* WkT    = (unsigned short*)(ws + OFF_WKT);
    unsigned short* WvT    = (unsigned short*)(ws + OFF_WVT);
    unsigned short* WoT    = (unsigned short*)(ws + OFF_WOT);
    unsigned short* Qb     = (unsigned short*)(ws + OFF_QB);
    unsigned short* Kb     = (unsigned short*)(ws + OFF_KB);
    unsigned short* Vb     = (unsigned short*)(ws + OFF_VB);
    unsigned short* ctxb   = (unsigned short*)(ws + OFF_CTX);

    // 1) weights -> bf16, transposed (WT[n][k]); Wq indexed by pathology on device
    cast_weights_kernel<<<dim3(DMODEL / 32, DMODEL / 32, 4), dim3(32, 8), 0, stream>>>(
        Wq, Wk, Wv, Wo, pidx, WqT, WkT, WvT, WoT);

    // 2) mean over S + bf16 cast of activations
    mean_cast_kernel<<<(BATCH * DMODEL) / 256, 256, 0, stream>>>(seq, seq_bf, q_bf);

    // 3) projections (bf16 WMMA GEMMs)
    gemm_bf16_wmma<false><<<dim3(DMODEL / BN, BATCH / BM), 256, 0, stream>>>(
        q_bf, WqT, bq, pidx, DMODEL, (void*)Qb);                       // Q = mean @ Wq[p] + bq[p]
    gemm_bf16_wmma<false><<<dim3(DMODEL / BN, (BATCH * SEQ) / BM), 256, 0, stream>>>(
        seq_bf, WkT, bk, nullptr, 0, (void*)Kb);                       // K
    gemm_bf16_wmma<false><<<dim3(DMODEL / BN, (BATCH * SEQ) / BM), 256, 0, stream>>>(
        seq_bf, WvT, bv, nullptr, 0, (void*)Vb);                       // V

    // 4) attention (wave per (b,h))
    attention_kernel<<<(BATCH * NHEAD) / 8, 256, 0, stream>>>(Qb, Kb, Vb, avail, ctxb);

    // 5) output projection straight into d_out (f32), then LayerNorm in place
    gemm_bf16_wmma<true><<<dim3(DMODEL / BN, BATCH / BM), 256, 0, stream>>>(
        ctxb, WoT, bo, nullptr, 0, (void*)out);
    layernorm_kernel<<<BATCH, 256, 0, stream>>>(out, gamma, beta);
}
